// GRUController_36112085025199
// MI455X (gfx1250) — matrix operand order repaired and checked by reading, not compiled, not run on hardware
//
#include <hip/hip_runtime.h>

// Fused GRU encoder(2L, T=256) + autoregressive decoder(2L, 180) + head.
// One persistent kernel: block = 16 batch rows, 128 threads = 4 waves.
// Recurrent matmuls via v_wmma_f32_16x16x32_f16; hidden state lives in LDS.
// Gate nonlinearities use raw v_rcp_f32 / v_exp_f32 (TRANS32, co-executes with VALU)
// instead of IEEE div / libm tanh -- the recurrence is latency-bound.

typedef __attribute__((ext_vector_type(16))) _Float16 v16h;
typedef __attribute__((ext_vector_type(8)))  float    v8f;

#define H    64
#define G3   192
#define LD   72    // padded LDS row stride (halves) for h buffers: 144B -> conflict-free
#define LDW  72    // padded LDS row stride for staged Wih1
#define TSEQ 256
#define NIN  6

#define LOG2E 1.4426950408889634f

__device__ __forceinline__ float fast_sigmoid(float x) {
  // 1/(1+e^-x) via v_exp_f32 + v_rcp_f32 (no div fixup chain)
  return __builtin_amdgcn_rcpf(1.f + __builtin_amdgcn_exp2f(-LOG2E * x));
}
__device__ __forceinline__ float fast_tanh(float x) {
  // tanh(x) = 1 - 2/(e^(2x)+1)
  return 1.f - 2.f * __builtin_amdgcn_rcpf(1.f + __builtin_amdgcn_exp2f((2.f * LOG2E) * x));
}

// ---- A fragment (16x32 f16, M x K) from LDS h buffer ----
// lane<16: M=lane, K = [0..15] (+32*kHalf); lane>=16: M=lane-16, K=[16..31]
__device__ __forceinline__ v16h load_afrag(const _Float16* buf, int lane, int kHalf) {
  int m  = lane & 15;
  int k0 = kHalf * 32 + ((lane >> 4) << 4);
  const _Float16* p = buf + m * LD + k0;
  v16h a;
#pragma unroll
  for (int i = 0; i < 8; ++i) { a[i] = (_Float16)p[i]; a[i + 8] = (_Float16)p[i + 8]; }
  return a;
}

// ---- B fragment (32x16 f16, K x N) : B[k][n] = W[row=c][k], W row-major [192][64] ----
// lane holds col N=lane&15; lanes<16 hold K kbase..kbase+15, lanes>=16 K kbase+16.. (kbase folds that in)
__device__ __forceinline__ v16h load_bfrag_g(const float* W, int c, int kbase) {
  const float* p = W + c * H + kbase;
  v16h b;
#pragma unroll
  for (int i = 0; i < 16; ++i) b[i] = (_Float16)p[i];
  return b;
}

__device__ __forceinline__ v16h load_bfrag_lds(const _Float16* W, int row, int kbase) {
  const _Float16* p = W + row * LDW + kbase;
  v16h b;
#pragma unroll
  for (int i = 0; i < 16; ++i) b[i] = (_Float16)p[i];
  return b;
}

__device__ __forceinline__ v8f wmma_acc(v16h a, v16h b, v8f c) {
  return __builtin_amdgcn_wmma_f32_16x16x32_f16(false, a, false, b, (short)0, c, false, false);
}

// Gate math + h update for this wave's 16x16 chunk. C-layout: elem e -> row m=e+8*(lane>=16), col c.
__device__ __forceinline__ void gru_combine(v8f aR, v8f aZ, v8f aI, v8f aN,
                                            float br, float bz, float bi, float bn,
                                            _Float16* hbuf, int hi8, int c) {
  float hn[8];
#pragma unroll
  for (int e = 0; e < 8; ++e) {
    int m   = e + hi8;
    float r = fast_sigmoid(aR[e] + br);
    float z = fast_sigmoid(aZ[e] + bz);
    float n = fast_tanh(aI[e] + bi + r * (aN[e] + bn));
    float ho = (float)hbuf[m * LD + c];
    hn[e] = (1.f - z) * n + z * ho;
  }
  __syncthreads();               // everyone done reading old h (A-frags + h_old)
#pragma unroll
  for (int e = 0; e < 8; ++e) hbuf[(e + hi8) * LD + c] = (_Float16)hn[e];
  __syncthreads();               // new h visible
}

// Layer-1 style cell: gi = inbuf @ Wih^T (WMMA, Wih staged f16 in LDS), gh = hbuf @ Whh^T (WMMA, regs)
__device__ __forceinline__ void layer1_cell(const _Float16* inbuf, _Float16* hbuf,
                                            const _Float16* wx, const v16h (&Bh)[3][2],
                                            float br, float bz, float bi, float bn,
                                            int lane, int hi8, int c, int kb0) {
  v16h xL = load_afrag(inbuf, lane, 0);
  v16h xH = load_afrag(inbuf, lane, 1);
  v16h hL = load_afrag(hbuf, lane, 0);
  v16h hH = load_afrag(hbuf, lane, 1);
  v8f aR = {}, aZ = {}, aI = {}, aN = {};
  aR = wmma_acc(xL, load_bfrag_lds(wx, c, kb0), aR);
  aR = wmma_acc(xH, load_bfrag_lds(wx, c, 32 + kb0), aR);
  aR = wmma_acc(hL, Bh[0][0], aR);
  aR = wmma_acc(hH, Bh[0][1], aR);
  aZ = wmma_acc(xL, load_bfrag_lds(wx, 64 + c, kb0), aZ);
  aZ = wmma_acc(xH, load_bfrag_lds(wx, 64 + c, 32 + kb0), aZ);
  aZ = wmma_acc(hL, Bh[1][0], aZ);
  aZ = wmma_acc(hH, Bh[1][1], aZ);
  aI = wmma_acc(xL, load_bfrag_lds(wx, 128 + c, kb0), aI);
  aI = wmma_acc(xH, load_bfrag_lds(wx, 128 + c, 32 + kb0), aI);
  aN = wmma_acc(hL, Bh[2][0], aN);
  aN = wmma_acc(hH, Bh[2][1], aN);
  gru_combine(aR, aZ, aI, aN, br, bz, bi, bn, hbuf, hi8, c);
}

__global__ __launch_bounds__(128, 1)
void gru_seq2seq_kernel(const float* __restrict__ x,
                        const float* __restrict__ eWih0, const float* __restrict__ eWhh0,
                        const float* __restrict__ ebih0, const float* __restrict__ ebhh0,
                        const float* __restrict__ eWih1, const float* __restrict__ eWhh1,
                        const float* __restrict__ ebih1, const float* __restrict__ ebhh1,
                        const float* __restrict__ dWih0, const float* __restrict__ dWhh0,
                        const float* __restrict__ dbih0, const float* __restrict__ dbhh0,
                        const float* __restrict__ dWih1, const float* __restrict__ dWhh1,
                        const float* __restrict__ dbih1, const float* __restrict__ dbhh1,
                        const float* __restrict__ outW, const float* __restrict__ outb,
                        const int* __restrict__ tlen,
                        float* __restrict__ out) {
  __shared__ __align__(16) _Float16 h0buf[16 * LD];
  __shared__ __align__(16) _Float16 h1buf[16 * LD];
  __shared__ __align__(16) _Float16 wxbuf[G3 * LDW];
  __shared__ float xbuf[16 * NIN];
  __shared__ float red[128];
  __shared__ float prevbuf[16];

  const int tid  = threadIdx.x;
  const int lane = tid & 31;
  const int wv   = tid >> 5;
  const int cn   = lane & 15;
  const int hi8  = (lane >> 4) << 3;          // 0 or 8: row offset of this lane's C elems
  const int c    = wv * 16 + cn;              // 0..63: column within each gate block
  const int kb0  = (lane >> 4) * 16;          // 0/16: B-frag K offset within a 32-chunk
  const int b0   = blockIdx.x * 16;

  for (int i = tid; i < 16 * LD; i += 128) { h0buf[i] = (_Float16)0.f; h1buf[i] = (_Float16)0.f; }

  // ---------------- encoder weights ----------------
  v16h Bh0[3][2], Bh1[3][2];
#pragma unroll
  for (int g = 0; g < 3; ++g)
#pragma unroll
    for (int kt = 0; kt < 2; ++kt) {
      Bh0[g][kt] = load_bfrag_g(eWhh0, g * 64 + c, kt * 32 + kb0);
      Bh1[g][kt] = load_bfrag_g(eWhh1, g * 64 + c, kt * 32 + kb0);
    }
  for (int i = tid; i < G3 * H; i += 128)
    wxbuf[(i >> 6) * LDW + (i & 63)] = (_Float16)eWih1[i];

  float br0 = ebih0[c] + ebhh0[c], bz0 = ebih0[64 + c] + ebhh0[64 + c];
  float bi0 = ebih0[128 + c],      bn0 = ebhh0[128 + c];
  float br1 = ebih1[c] + ebhh1[c], bz1 = ebih1[64 + c] + ebhh1[64 + c];
  float bi1 = ebih1[128 + c],      bn1 = ebhh1[128 + c];
  float wxr[NIN], wxz[NIN], wxn[NIN];
#pragma unroll
  for (int k = 0; k < NIN; ++k) {
    wxr[k] = eWih0[c * NIN + k];
    wxz[k] = eWih0[(64 + c) * NIN + k];
    wxn[k] = eWih0[(128 + c) * NIN + k];
  }
  __syncthreads();

  // ---------------- encoder time loop ----------------
  for (int t = 0; t < TSEQ; ++t) {
    if (tid < 16 * NIN)
      xbuf[tid] = x[(b0 + tid / NIN) * (TSEQ * NIN) + t * NIN + (tid % NIN)];
    __syncthreads();

    { // layer 0: gi elementwise (K=6), gh via WMMA
      v16h aL = load_afrag(h0buf, lane, 0);
      v16h aH = load_afrag(h0buf, lane, 1);
      v8f aR = {}, aZ = {}, aN = {};
      aR = wmma_acc(aL, Bh0[0][0], aR); aR = wmma_acc(aH, Bh0[0][1], aR);
      aZ = wmma_acc(aL, Bh0[1][0], aZ); aZ = wmma_acc(aH, Bh0[1][1], aZ);
      aN = wmma_acc(aL, Bh0[2][0], aN); aN = wmma_acc(aH, Bh0[2][1], aN);
      v8f aI = {};
#pragma unroll
      for (int e = 0; e < 8; ++e) {
        const float* xr = &xbuf[(e + hi8) * NIN];
        float gr = 0.f, gz = 0.f, gn = 0.f;
#pragma unroll
        for (int k = 0; k < NIN; ++k) { gr += xr[k] * wxr[k]; gz += xr[k] * wxz[k]; gn += xr[k] * wxn[k]; }
        aR[e] += gr; aZ[e] += gz; aI[e] = gn;
      }
      gru_combine(aR, aZ, aI, aN, br0, bz0, bi0, bn0, h0buf, hi8, c);
    }
    layer1_cell(h0buf, h1buf, wxbuf, Bh1, br1, bz1, bi1, bn1, lane, hi8, c, kb0);
  }

  // ---------------- decoder weights (reuse registers/LDS) ----------------
#pragma unroll
  for (int g = 0; g < 3; ++g)
#pragma unroll
    for (int kt = 0; kt < 2; ++kt) {
      Bh0[g][kt] = load_bfrag_g(dWhh0, g * 64 + c, kt * 32 + kb0);
      Bh1[g][kt] = load_bfrag_g(dWhh1, g * 64 + c, kt * 32 + kb0);
    }
  for (int i = tid; i < G3 * H; i += 128)
    wxbuf[(i >> 6) * LDW + (i & 63)] = (_Float16)dWih1[i];
  br0 = dbih0[c] + dbhh0[c]; bz0 = dbih0[64 + c] + dbhh0[64 + c];
  bi0 = dbih0[128 + c];      bn0 = dbhh0[128 + c];
  br1 = dbih1[c] + dbhh1[c]; bz1 = dbih1[64 + c] + dbhh1[64 + c];
  bi1 = dbih1[128 + c];      bn1 = dbhh1[128 + c];
  const float wdr = dWih0[c], wdz = dWih0[64 + c], wdn = dWih0[128 + c];
  float owc[8];
#pragma unroll
  for (int q = 0; q < 8; ++q) owc[q] = outW[(tid & 7) * 8 + q];
  const float ob = outb[0];
  if (tid < 16) prevbuf[tid] = 0.f;
  const int TL = tlen[0];
  __syncthreads();

  // ---------------- decoder time loop ----------------
  for (int s = 0; s < TL; ++s) {
    { // layer 0: gi = prev * Wih0 (K=1), gh via WMMA; h0/h1 carry over from encoder finals
      v16h aL = load_afrag(h0buf, lane, 0);
      v16h aH = load_afrag(h0buf, lane, 1);
      v8f aR = {}, aZ = {}, aN = {};
      aR = wmma_acc(aL, Bh0[0][0], aR); aR = wmma_acc(aH, Bh0[0][1], aR);
      aZ = wmma_acc(aL, Bh0[1][0], aZ); aZ = wmma_acc(aH, Bh0[1][1], aZ);
      aN = wmma_acc(aL, Bh0[2][0], aN); aN = wmma_acc(aH, Bh0[2][1], aN);
      v8f aI = {};
#pragma unroll
      for (int e = 0; e < 8; ++e) {
        float p = prevbuf[e + hi8];
        aR[e] += p * wdr; aZ[e] += p * wdz; aI[e] = p * wdn;
      }
      gru_combine(aR, aZ, aI, aN, br0, bz0, bi0, bn0, h0buf, hi8, c);
    }
    layer1_cell(h0buf, h1buf, wxbuf, Bh1, br1, bz1, bi1, bn1, lane, hi8, c, kb0);

    { // head: cv[m] = out_b + sum_j h1[m][j]*outW[j]; LDS tree reduce, feed back as prev
      int m = tid >> 3, seg = tid & 7;
      float p = 0.f;
#pragma unroll
      for (int q = 0; q < 8; ++q) p += (float)h1buf[m * LD + seg * 8 + q] * owc[q];
      red[tid] = p;
      __syncthreads();
      if (tid < 16) {
        float cv = ob;
#pragma unroll
        for (int q = 0; q < 8; ++q) cv += red[tid * 8 + q];
        out[(b0 + tid) * TL + s] = cv;
        prevbuf[tid] = cv;
      }
      __syncthreads();
    }
  }
}

extern "C" void kernel_launch(void* const* d_in, const int* in_sizes, int n_in,
                              void* d_out, int out_size, void* d_ws, size_t ws_size,
                              hipStream_t stream) {
  (void)n_in; (void)d_ws; (void)ws_size; (void)out_size;
  const float* xp = (const float*)d_in[0];
  const int B = in_sizes[0] / (TSEQ * NIN);   // 4096
  dim3 grid(B / 16), block(128);
  gru_seq2seq_kernel<<<grid, block, 0, stream>>>(
      xp,
      (const float*)d_in[1],  (const float*)d_in[2],  (const float*)d_in[3],  (const float*)d_in[4],
      (const float*)d_in[5],  (const float*)d_in[6],  (const float*)d_in[7],  (const float*)d_in[8],
      (const float*)d_in[9],  (const float*)d_in[10], (const float*)d_in[11], (const float*)d_in[12],
      (const float*)d_in[13], (const float*)d_in[14], (const float*)d_in[15], (const float*)d_in[16],
      (const float*)d_in[17], (const float*)d_in[18],
      (const int*)d_in[19],
      (float*)d_out);
}
